// STN3d_90494960927417
// MI455X (gfx1250) — matrix operand order repaired and checked by reading, not compiled, not run on hardware
//
#include <hip/hip_runtime.h>

typedef _Float16 half_t;
typedef _Float16 v16h __attribute__((ext_vector_type(16)));
typedef float     v8f  __attribute__((ext_vector_type(8)));

#define P_TOT 32768     // B(16) * N(2048) points
#define EPS_BN 1e-5f
#define BINS 73
#define STAT_ROWS 128

// ---------------- utility kernels ----------------

__global__ void zero_f32(float* p, int n) {
  int t = blockIdx.x * 256 + threadIdx.x;
  if (t < n) p[t] = 0.f;
}

__global__ void cvt_f16(const float* __restrict__ s, half_t* __restrict__ d, int n) {
  int t = blockIdx.x * 256 + threadIdx.x;
  if (t < n) d[t] = (half_t)s[t];
}

// ---------------- conv1: 3 -> 64 (K tiny, VALU) ----------------
// x: [16, 3, 2048] f32; out z1: [P, 64] f16 (point-major)
__global__ void conv1_kernel(const float* __restrict__ x, const float* __restrict__ w,
                             const float* __restrict__ bias, half_t* __restrict__ z) {
  int t  = blockIdx.x * 256 + threadIdx.x;   // P*64 threads
  int co = t & 63;
  int p  = t >> 6;
  int b  = p >> 11, n = p & 2047;
  const float* xb = x + (size_t)b * 3 * 2048 + n;
  float v = bias[co] + w[co*3 + 0] * xb[0]
                     + w[co*3 + 1] * xb[2048]
                     + w[co*3 + 2] * xb[4096];
  z[(size_t)p * 64 + co] = (half_t)v;
}

// ---------------- per-channel sum / sumsq over all P points ----------------
// z: [P, C] f16, C multiple of 64
__global__ void stats_kernel(const half_t* __restrict__ z, float* __restrict__ sum,
                             float* __restrict__ sumsq, int C) {
  __shared__ float ss[64], sq[64];
  int t  = threadIdx.x;                 // 256
  int cl = t & 63;
  int c  = blockIdx.y * 64 + cl;
  int p0 = blockIdx.x * STAT_ROWS;
  float s = 0.f, q = 0.f;
  for (int r = (t >> 6); r < STAT_ROWS; r += 4) {
    float v = (float)z[(size_t)(p0 + r) * C + c];
    s += v; q += v * v;
  }
  if (t < 64) { ss[t] = 0.f; sq[t] = 0.f; }
  __syncthreads();
  atomicAdd(&ss[cl], s);
  atomicAdd(&sq[cl], q);
  __syncthreads();
  if (t < 64) {
    atomicAdd(&sum[c],   ss[t]);
    atomicAdd(&sumsq[c], sq[t]);
  }
}

// in-place BN (training mode, biased var over B*N) + ReLU; cmask = C-1 (C pow2)
__global__ void bn_relu_conv(half_t* z, const float* __restrict__ sum,
                             const float* __restrict__ sumsq,
                             const float* __restrict__ g, const float* __restrict__ be,
                             int cmask) {
  size_t t = (size_t)blockIdx.x * 256 + threadIdx.x;   // t = p*C + c
  int c = (int)t & cmask;
  float m   = sum[c]   * (1.f / (float)P_TOT);
  float var = sumsq[c] * (1.f / (float)P_TOT) - m * m;
  float sc  = g[c] * rsqrtf(var + EPS_BN);
  float sh  = be[c] - m * sc;
  float v   = (float)z[t] * sc + sh;
  z[t] = (half_t)(v > 0.f ? v : 0.f);
}

// ---------------- generic WMMA f16 GEMM, 4 co-tiles per wave ----------------
// Zout[p][co] = sum_k Wt[co][k] * Hin[p][k] + bias[co]
// Hin: [P, CI] f16, Wt: [CO, CI] f16, Zout: [P, CO] f16.
// CI mult of 64, CO mult of 64, P mult of 16.
// Each wave: one 16-wide p-tile x 64 output channels (4 WMMA acc tiles).
// B fragment reused across the 4 WMMAs per K-step; explicit double buffering
// so next K-step's loads are in flight while current WMMAs issue.
__global__ void __launch_bounds__(256)
gemm_wmma(const half_t* __restrict__ Hin, const half_t* __restrict__ Wt,
          const float* __restrict__ bias, half_t* __restrict__ Zout,
          int P, int CI, int CO) {
  int lane    = threadIdx.x & 31;
  int wavegrp = blockIdx.x * 8 + (threadIdx.x >> 5);
  int ptiles  = P >> 4;
  int ngroups = ptiles * (CO >> 6);
  if (wavegrp >= ngroups) return;           // wave-uniform; EXEC stays all-ones
  int p0  = (wavegrp % ptiles) << 4;
  int co0 = (wavegrp / ptiles) << 6;        // 64 output channels per wave
  int m  = lane & 15;                       // A: row M=co; B/D: col N=p
  int hi = lane >> 4;

  // B (32x16 f16 per K-step): 16 contiguous halves at K = 16*hi
  const half_t* hrow = Hin + (size_t)(p0 + m) * CI + 16 * hi;
  // A (16x32 f16): 16 contiguous halves at K = 16*hi; co-tile g at +16*g rows
  const half_t* wrow = Wt + (size_t)(co0 + m) * CI + 16 * hi;
  size_t wstride = (size_t)16 * CI;         // next co-tile

  v8f acc[4];
#pragma unroll
  for (int g = 0; g < 4; ++g) acc[g] = {0.f, 0.f, 0.f, 0.f, 0.f, 0.f, 0.f, 0.f};

  // prologue: load K-step 0
  v16h bcur, acur[4];
  ((uint4*)&bcur)[0] = *(const uint4*)(hrow);
  ((uint4*)&bcur)[1] = *(const uint4*)(hrow + 8);
#pragma unroll
  for (int g = 0; g < 4; ++g) {
    const half_t* wg = wrow + (size_t)g * wstride;
    ((uint4*)&acur[g])[0] = *(const uint4*)(wg);
    ((uint4*)&acur[g])[1] = *(const uint4*)(wg + 8);
  }

  for (int k = 32; k < CI; k += 32) {
    // prefetch next K-step while current WMMAs issue
    v16h bnxt, anxt[4];
    ((uint4*)&bnxt)[0] = *(const uint4*)(hrow + k);
    ((uint4*)&bnxt)[1] = *(const uint4*)(hrow + k + 8);
#pragma unroll
    for (int g = 0; g < 4; ++g) {
      const half_t* wg = wrow + (size_t)g * wstride + k;
      ((uint4*)&anxt[g])[0] = *(const uint4*)(wg);
      ((uint4*)&anxt[g])[1] = *(const uint4*)(wg + 8);
    }
#pragma unroll
    for (int g = 0; g < 4; ++g)
      acc[g] = __builtin_amdgcn_wmma_f32_16x16x32_f16(false, acur[g], false, bcur,
                                                      (short)0, acc[g], false, false);
    bcur = bnxt;
#pragma unroll
    for (int g = 0; g < 4; ++g) acur[g] = anxt[g];
  }
#pragma unroll
  for (int g = 0; g < 4; ++g)
    acc[g] = __builtin_amdgcn_wmma_f32_16x16x32_f16(false, acur[g], false, bcur,
                                                    (short)0, acc[g], false, false);

  // D: lane&15 = p column; VGPR r -> co = cotile + r + 8*hi (8 contiguous f16/lane)
  half_t* zrow = Zout + (size_t)(p0 + m) * CO + co0 + 8 * hi;
#pragma unroll
  for (int g = 0; g < 4; ++g) {
    int cb = co0 + 16 * g + 8 * hi;
    half_t* zr = zrow + 16 * g;
#pragma unroll
    for (int r = 0; r < 8; ++r)
      zr[r] = (half_t)(acc[g][r] + bias[cb + r]);
  }
}

// ---------------- histogram(73 bins on [-10,10]) + argmax pooling ----------------
// h3: [P, 1024] f16, p = b*2048 + n. One block per (b, 64-feature chunk).
__global__ void __launch_bounds__(256)
hist_kernel(const half_t* __restrict__ h3, half_t* __restrict__ feat) {
  __shared__ unsigned hist[64 * BINS];
  int t  = threadIdx.x;
  int b  = blockIdx.x >> 4;
  int f0 = (blockIdx.x & 15) << 6;
  for (int i = t; i < 64 * BINS; i += 256) hist[i] = 0u;
  __syncthreads();
  int fl = t & 63;
  const half_t* base = h3 + ((size_t)b * 2048) * 1024 + f0 + fl;
  for (int r = (t >> 6); r < 2048; r += 4) {
    float v = (float)base[(size_t)r * 1024];
    if (v >= -10.f && v <= 10.f) {
      int bin = (int)((v + 10.f) * (73.f / 20.f));
      if (bin > 72) bin = 72;
      atomicAdd(&hist[fl * BINS + bin], 1u);
    }
  }
  __syncthreads();
  if (t < 64) {
    unsigned best = hist[t * BINS];
    int bi = 0;
    for (int k = 1; k < BINS; ++k) {
      unsigned cnt = hist[t * BINS + k];
      if (cnt > best) { best = cnt; bi = k; }   // first max, like argmax
    }
    feat[(size_t)b * 1024 + f0 + t] = (half_t)bi;
  }
}

// ---------------- FC batch-norm (over batch of 16) + ReLU, in place ----------------
__global__ void bn_relu_fc(half_t* z, const float* __restrict__ g,
                           const float* __restrict__ be, int C) {
  int c = blockIdx.x * 256 + threadIdx.x;
  if (c >= C) return;
  float s = 0.f, q = 0.f;
  for (int b = 0; b < 16; ++b) { float v = (float)z[b * C + c]; s += v; q += v * v; }
  float m   = s * (1.f / 16.f);
  float var = q * (1.f / 16.f) - m * m;
  float sc  = g[c] * rsqrtf(var + EPS_BN);
  float sh  = be[c] - m * sc;
  for (int b = 0; b < 16; ++b) {
    float v = (float)z[b * C + c] * sc + sh;
    z[b * C + c] = (half_t)(v > 0.f ? v : 0.f);
  }
}

// ---------------- FC3: 256 -> 9, add identity, f32 out [16,3,3] ----------------
__global__ void fc3_kernel(const half_t* __restrict__ h, const float* __restrict__ w,
                           const float* __restrict__ bias, float* __restrict__ out) {
  int t = threadIdx.x;
  if (t >= 144) return;
  int b = t / 9, o = t % 9;
  float s = bias[o];
  for (int k = 0; k < 256; ++k) s += (float)h[b * 256 + k] * w[o * 256 + k];
  if (o == 0 || o == 4 || o == 8) s += 1.f;
  out[b * 9 + o] = s;
}

// ---------------- host-side launch ----------------

extern "C" void kernel_launch(void* const* d_in, const int* in_sizes, int n_in,
                              void* d_out, int out_size, void* d_ws, size_t ws_size,
                              hipStream_t stream) {
  (void)in_sizes; (void)n_in; (void)out_size; (void)ws_size;
  const float* x   = (const float*)d_in[0];
  const float* w1  = (const float*)d_in[1];
  const float* b1  = (const float*)d_in[2];
  const float* g1  = (const float*)d_in[3];
  const float* be1 = (const float*)d_in[4];
  const float* w2  = (const float*)d_in[5];
  const float* b2  = (const float*)d_in[6];
  const float* g2  = (const float*)d_in[7];
  const float* be2 = (const float*)d_in[8];
  const float* w3  = (const float*)d_in[9];
  const float* b3  = (const float*)d_in[10];
  const float* g3  = (const float*)d_in[11];
  const float* be3 = (const float*)d_in[12];
  const float* fw1 = (const float*)d_in[13];
  const float* fb1 = (const float*)d_in[14];
  const float* g4  = (const float*)d_in[15];
  const float* be4 = (const float*)d_in[16];
  const float* fw2 = (const float*)d_in[17];
  const float* fb2 = (const float*)d_in[18];
  const float* g5  = (const float*)d_in[19];
  const float* be5 = (const float*)d_in[20];
  const float* fw3 = (const float*)d_in[21];
  const float* fb3 = (const float*)d_in[22];
  float* out = (float*)d_out;

  char*  ws  = (char*)d_ws;
  size_t off = 0;
  auto carve = [&](size_t bytes) -> char* {
    char* p = ws + off;
    off += (bytes + 255) & ~(size_t)255;
    return p;
  };

  float*  stats = (float*)carve(2432 * sizeof(float));   // sums+sumsqs for 64+128+1024 ch
  float *sum1 = stats,       *sq1 = stats + 64;
  float *sum2 = stats + 192, *sq2 = stats + 320;
  float *sum3 = stats + 448, *sq3 = stats + 1472;
  half_t* w2h  = (half_t*)carve((size_t)128 * 64 * 2);
  half_t* w3h  = (half_t*)carve((size_t)1024 * 128 * 2);
  half_t* fw1h = (half_t*)carve((size_t)512 * 1024 * 2);
  half_t* fw2h = (half_t*)carve((size_t)256 * 512 * 2);
  half_t* z1   = (half_t*)carve((size_t)P_TOT * 64 * 2);     // 4 MB
  half_t* z2   = (half_t*)carve((size_t)P_TOT * 128 * 2);    // 8 MB
  half_t* z3   = (half_t*)carve((size_t)P_TOT * 1024 * 2);   // 64 MB
  half_t* feat = (half_t*)carve((size_t)16 * 1024 * 2);
  half_t* zf1  = (half_t*)carve((size_t)16 * 512 * 2);
  half_t* zf2  = (half_t*)carve((size_t)16 * 256 * 2);

  zero_f32<<<(2432 + 255) / 256, 256, 0, stream>>>(stats, 2432);
  cvt_f16<<<(128 * 64 + 255) / 256, 256, 0, stream>>>(w2, w2h, 128 * 64);
  cvt_f16<<<(1024 * 128 + 255) / 256, 256, 0, stream>>>(w3, w3h, 1024 * 128);
  cvt_f16<<<(512 * 1024 + 255) / 256, 256, 0, stream>>>(fw1, fw1h, 512 * 1024);
  cvt_f16<<<(256 * 512 + 255) / 256, 256, 0, stream>>>(fw2, fw2h, 256 * 512);

  // layer 1
  conv1_kernel<<<P_TOT * 64 / 256, 256, 0, stream>>>(x, w1, b1, z1);
  stats_kernel<<<dim3(P_TOT / STAT_ROWS, 1), 256, 0, stream>>>(z1, sum1, sq1, 64);
  bn_relu_conv<<<P_TOT * 64 / 256, 256, 0, stream>>>(z1, sum1, sq1, g1, be1, 63);

  // layer 2: 64 -> 128   (2048 p-tiles * 2 co-groups = 4096 waves)
  gemm_wmma<<<512, 256, 0, stream>>>(z1, w2h, b2, z2, P_TOT, 64, 128);
  stats_kernel<<<dim3(P_TOT / STAT_ROWS, 2), 256, 0, stream>>>(z2, sum2, sq2, 128);
  bn_relu_conv<<<P_TOT * 128 / 256, 256, 0, stream>>>(z2, sum2, sq2, g2, be2, 127);

  // layer 3: 128 -> 1024 (2048 p-tiles * 16 co-groups = 32768 waves)
  gemm_wmma<<<4096, 256, 0, stream>>>(z2, w3h, b3, z3, P_TOT, 128, 1024);
  stats_kernel<<<dim3(P_TOT / STAT_ROWS, 16), 256, 0, stream>>>(z3, sum3, sq3, 1024);
  bn_relu_conv<<<(P_TOT / 256) * 1024, 256, 0, stream>>>(z3, sum3, sq3, g3, be3, 1023);

  // histogram argmax pooling -> feat [16,1024]
  hist_kernel<<<256, 256, 0, stream>>>(z3, feat);

  // FC head (batch of 16 == one WMMA tile dimension)
  gemm_wmma<<<1, 256, 0, stream>>>(feat, fw1h, fb1, zf1, 16, 1024, 512);   // 8 waves
  bn_relu_fc<<<2, 256, 0, stream>>>(zf1, g4, be4, 512);
  gemm_wmma<<<1, 256, 0, stream>>>(zf1, fw2h, fb2, zf2, 16, 512, 256);     // 4 waves
  bn_relu_fc<<<1, 256, 0, stream>>>(zf2, g5, be5, 256);
  fc3_kernel<<<1, 160, 0, stream>>>(zf2, fw3, fb3, out);
}
